// PDAN_50663434224530
// MI455X (gfx1250) — matrix operand order repaired and checked by reading, not compiled, not run on hardware
//
#include <hip/hip_runtime.h>
#include <hip/hip_bf16.h>

// ---------------------------------------------------------------------------
// PDAN forward for MI455X (gfx1250).
// Heavy math = bf16 WMMA GEMMs (v_wmma_f32_16x16x32_bf16), fp32 epilogue.
// - LDS tiles pre-swizzled per WMMA lane layout: every fragment load is two
//   ds_load_b128, every staging write is a wide ds_store (b128/b64).
// - Double-buffered LDS: one barrier per K-step; next chunk staged while the
//   current chunk's 8 WMMAs run.
// - Each wave owns a 32x64 tile: 2 A-frags x 4 B-frags -> 8 WMMAs per K-step.
// ---------------------------------------------------------------------------

typedef __attribute__((ext_vector_type(16))) __bf16 v16bf;
typedef __attribute__((ext_vector_type(8)))  __bf16 v8bf;
typedef __attribute__((ext_vector_type(4)))  __bf16 v4bf;
typedef __attribute__((ext_vector_type(8)))  float  v8f;

__device__ __forceinline__ __bf16 f2bf(float f) { return (__bf16)f; }

#define BM 128
#define BN 128
#define KC 32
#define CHUNK 24   // 16 payload bf16 + 8 pad: 48B stride = 12-bank rotation,
                   // keeps 16B alignment for ds_load_b128 / ds_store_b128
#define ABUF (2 * BM * CHUNK)   // one A buffer (bf16 elems)
#define BBUF (2 * BN * CHUNK)   // one B buffer

// A lane (h=lane>>4, m=lane&15) wants K = {8h..8h+7} u {16+8h..16+8h+7}.
// B lane (h=lane>>4, n=lane&15) wants K = {8h..8h+7} u {16+8h..16+8h+7}.
// Both stored contiguously at chunk (h*ROWS + row)*CHUNK.
__device__ __forceinline__ v16bf load_frag(const __bf16* lds, int rows,
                                           int lane, int r0) {
    int r = r0 + (lane & 15);
    int h = lane >> 4;
    const __bf16* p = lds + (h * rows + r) * CHUNK;
    v8bf lo = *(const v8bf*)(p);
    v8bf hi = *(const v8bf*)(p + 8);
    return __builtin_shufflevector(lo, hi, 0, 1, 2, 3, 4, 5, 6, 7,
                                            8, 9, 10, 11, 12, 13, 14, 15);
}

// Stage W chunk (BM x KC) -> swizzled LDS: 4 float4 loads, 2 ds_store_b128.
// k in [0,7]->h0 e0..7; [8,15]->h1 e0..7; [16,23]->h0 e8..15; [24,31]->h1 e8..15
__device__ __forceinline__ void stage_W(__bf16* lA, const float* __restrict__ W,
                                        int O, int I, int gM, int kc, int tid) {
    int idx = tid * 16;            // BM*KC = 4096 elems / 256 threads
    int m   = idx >> 5;            // 0..127
    int kk  = idx & 31;            // 0 or 16
    int e0  = kk >> 1;             // 0 or 8
    int go  = gM + m;
    float4 w0 = make_float4(0.f, 0.f, 0.f, 0.f), w1 = w0, w2 = w0, w3 = w0;
    if (go < O) {
        const float* wp = W + (size_t)go * I + kc + kk;
        w0 = *(const float4*)(wp);
        w1 = *(const float4*)(wp + 4);
        w2 = *(const float4*)(wp + 8);
        w3 = *(const float4*)(wp + 12);
    }
    v8bf v;
    v[0] = f2bf(w0.x); v[1] = f2bf(w0.y); v[2] = f2bf(w0.z); v[3] = f2bf(w0.w);
    v[4] = f2bf(w1.x); v[5] = f2bf(w1.y); v[6] = f2bf(w1.z); v[7] = f2bf(w1.w);
    *(v8bf*)(&lA[(0 * BM + m) * CHUNK + e0]) = v;          // h = 0
    v8bf u;
    u[0] = f2bf(w2.x); u[1] = f2bf(w2.y); u[2] = f2bf(w2.z); u[3] = f2bf(w2.w);
    u[4] = f2bf(w3.x); u[5] = f2bf(w3.y); u[6] = f2bf(w3.z); u[7] = f2bf(w3.w);
    *(v8bf*)(&lA[(1 * BM + m) * CHUNK + e0]) = u;          // h = 1
}

// Stage X chunk (KC x BN) -> swizzled LDS: 4 float4 loads, 4 ds_store_b64.
__device__ __forceinline__ void stage_X(__bf16* lB, const float* __restrict__ X,
                                        size_t xbase, int T, int kc, int tid) {
    int n0 = (tid & 31) * 4;       // column group
    int k0 = (tid >> 5) * 4;       // 0,4,..,28
    int h  = (k0 >> 3) & 1;
    int e  = (k0 & 7) + ((k0 >> 4) << 3);
    const float* xp = X + xbase + (size_t)(kc + k0) * T + n0;
    float4 r0 = *(const float4*)(xp);
    float4 r1 = *(const float4*)(xp + T);
    float4 r2 = *(const float4*)(xp + 2 * (size_t)T);
    float4 r3 = *(const float4*)(xp + 3 * (size_t)T);
#define STORE_COL(u, a, bb, c, d)                                            \
    { v4bf t; t[0] = f2bf(a); t[1] = f2bf(bb); t[2] = f2bf(c);               \
      t[3] = f2bf(d);                                                        \
      *(v4bf*)(&lB[(h * BN + n0 + (u)) * CHUNK + e]) = t; }
    STORE_COL(0, r0.x, r1.x, r2.x, r3.x)
    STORE_COL(1, r0.y, r1.y, r2.y, r3.y)
    STORE_COL(2, r0.z, r1.z, r2.z, r3.z)
    STORE_COL(3, r0.w, r1.w, r2.w, r3.w)
#undef STORE_COL
}

// Y[b,o,t] = sum_i W[o,i] * X[b,i,t]  (+bias[o]) (+res[b,o,t]) (*mask[b,t])
__global__ __launch_bounds__(256)
void gemm_wmma_bf16(const float* __restrict__ W, const float* __restrict__ X,
                    const float* __restrict__ bias, const float* __restrict__ res,
                    const float* __restrict__ mask, float* __restrict__ Y,
                    int O, int I, int T)
{
    __shared__ __bf16 lA[2 * ABUF];   // 24 KB (double buffered)
    __shared__ __bf16 lB[2 * BBUF];   // 24 KB

    const int tid  = threadIdx.x;
    const int lane = tid & 31;
    const int wv   = tid >> 5;           // wave id 0..7 (wave32)
    const int gT = blockIdx.x * BN;
    const int gM = blockIdx.y * BM;
    const int b  = blockIdx.z;

    // each wave owns a 32x64 tile: 2 A-frags x 4 B-frags -> 8 WMMAs/K-step
    const int m2 = (wv & 1) * 32;
    const int n4 = (wv >> 1) * 64;

    v8f acc[2][4] = {};

    const size_t xbase = (size_t)b * I * T + gT;

    // prologue: stage first K-chunk into buffer 0
    stage_W(lA, W, O, I, gM, 0, tid);
    stage_X(lB, X, xbase, T, 0, tid);

    int p = 0;
    for (int kc = 0; kc < I; kc += KC) {
        __syncthreads();   // buffer p staged for everyone

        // stage next chunk into buffer p^1 while buffer p feeds the XDL pipe
        if (kc + KC < I) {
            stage_W(lA + (p ^ 1) * ABUF, W, O, I, gM, kc + KC, tid);
            stage_X(lB + (p ^ 1) * BBUF, X, xbase, T, kc + KC, tid);
            if (kc + 2 * KC < I) {   // global_prefetch_b8 two chunks ahead
                __builtin_prefetch(W + (size_t)(gM + (tid >> 1)) * I + kc + 2 * KC, 0, 1);
                __builtin_prefetch(X + xbase + (size_t)(kc + 2 * KC + (tid & 31)) * T, 0, 1);
            }
        }

        const __bf16* cA = lA + p * ABUF;
        const __bf16* cB = lB + p * BBUF;
        v16bf a0 = load_frag(cA, BM, lane, m2);
        v16bf a1 = load_frag(cA, BM, lane, m2 + 16);
        v16bf b0 = load_frag(cB, BN, lane, n4);
        v16bf b1 = load_frag(cB, BN, lane, n4 + 16);
        v16bf b2 = load_frag(cB, BN, lane, n4 + 32);
        v16bf b3 = load_frag(cB, BN, lane, n4 + 48);

        acc[0][0] = __builtin_amdgcn_wmma_f32_16x16x32_bf16(false, a0, false, b0,
                                                            (short)0, acc[0][0], false, false);
        acc[0][1] = __builtin_amdgcn_wmma_f32_16x16x32_bf16(false, a0, false, b1,
                                                            (short)0, acc[0][1], false, false);
        acc[0][2] = __builtin_amdgcn_wmma_f32_16x16x32_bf16(false, a0, false, b2,
                                                            (short)0, acc[0][2], false, false);
        acc[0][3] = __builtin_amdgcn_wmma_f32_16x16x32_bf16(false, a0, false, b3,
                                                            (short)0, acc[0][3], false, false);
        acc[1][0] = __builtin_amdgcn_wmma_f32_16x16x32_bf16(false, a1, false, b0,
                                                            (short)0, acc[1][0], false, false);
        acc[1][1] = __builtin_amdgcn_wmma_f32_16x16x32_bf16(false, a1, false, b1,
                                                            (short)0, acc[1][1], false, false);
        acc[1][2] = __builtin_amdgcn_wmma_f32_16x16x32_bf16(false, a1, false, b2,
                                                            (short)0, acc[1][2], false, false);
        acc[1][3] = __builtin_amdgcn_wmma_f32_16x16x32_bf16(false, a1, false, b3,
                                                            (short)0, acc[1][3], false, false);
        p ^= 1;
    }

    // ---- epilogue: VGPR r holds M = r + 8*(lane>>4), N = lane&15 ----
    const int hh = lane >> 4;
    const int ln = lane & 15;
#pragma unroll
    for (int mi = 0; mi < 2; ++mi) {
#pragma unroll
        for (int nj = 0; nj < 4; ++nj) {
            int oBase = gM + m2 + mi * 16 + hh * 8;
            int t     = gT + n4 + nj * 16 + ln;
            const v8f& a = acc[mi][nj];
#pragma unroll
            for (int r = 0; r < 8; ++r) {
                int o = oBase + r;
                if (o < O) {
                    float y = a[r];
                    if (bias) y += bias[o];
                    size_t oidx = ((size_t)b * O + o) * T + t;
                    if (res)  y += res[oidx];
                    if (mask) y *= mask[(size_t)b * T + t];
                    Y[oidx] = y;
                }
            }
        }
    }
}

// Elementwise 3-tap dilated attention + ReLU (bandwidth bound).
// k_j[t] = ko[t+(j-1)d], v_j[t] = vo[t+(j-1)d], zero outside [0,T).
__global__ __launch_bounds__(256)
void dal_attn_relu(const float* __restrict__ q, const float* __restrict__ k,
                   const float* __restrict__ v, float* __restrict__ h,
                   int d, int T, long long total)
{
    long long i = (long long)blockIdx.x * blockDim.x + threadIdx.x;
    if (i >= total) return;
    int t = (int)(i % T);
    float qv = q[i];
    float km = (t >= d)    ? k[i - d] : 0.f;
    float kc = k[i];
    float kp = (t + d < T) ? k[i + d] : 0.f;
    float vm = (t >= d)    ? v[i - d] : 0.f;
    float vc = v[i];
    float vp = (t + d < T) ? v[i + d] : 0.f;
    float s0 = qv * km, s1 = qv * kc, s2 = qv * kp;
    float mx = fmaxf(s0, fmaxf(s1, s2));
    float e0 = __expf(s0 - mx), e1 = __expf(s1 - mx), e2 = __expf(s2 - mx);
    float o  = (e0 * vm + e1 * vc + e2 * vp) / (e0 + e1 + e2);
    h[i] = fmaxf(o, 0.f);
}

extern "C" void kernel_launch(void* const* d_in, const int* in_sizes, int n_in,
                              void* d_out, int out_size, void* d_ws, size_t ws_size,
                              hipStream_t stream)
{
    (void)in_sizes; (void)n_in; (void)out_size; (void)ws_size;

    const int B = 8, T = 4096, DIN = 1024, DM = 256, NC = 157;
    const int dils[5] = {1, 2, 4, 8, 16};

    const float* x    = (const float*)d_in[0];
    const float* mask = (const float*)d_in[1];
    const float* b0w  = (const float*)d_in[2];
    const float* b0b  = (const float*)d_in[3];
    const float* b6w  = (const float*)d_in[29];
    const float* b6b  = (const float*)d_in[30];

    const size_t N1 = (size_t)B * DM * T;          // 8.4M floats = 32 MB
    float* out = (float*)d_ws;
    float* qb  = out + N1;
    float* kb  = qb  + N1;
    float* vb  = kb  + N1;
    float* hb  = vb  + N1;

    dim3 blk(256);
    const int nBlk = T / BN;                       // 32

    // bottle0: DM x DIN GEMM + bias
    gemm_wmma_bf16<<<dim3(nBlk, (DM + BM - 1) / BM, B), blk, 0, stream>>>(
        b0w, x, b0b, nullptr, nullptr, out, DM, DIN, T);

    for (int l = 0; l < 5; ++l) {
        const float* wq = (const float*)d_in[4 + l * 5 + 0];
        const float* wk = (const float*)d_in[4 + l * 5 + 1];
        const float* wv = (const float*)d_in[4 + l * 5 + 2];
        const float* bw = (const float*)d_in[4 + l * 5 + 3];
        const float* bb = (const float*)d_in[4 + l * 5 + 4];

        gemm_wmma_bf16<<<dim3(nBlk, DM / BM, B), blk, 0, stream>>>(
            wq, out, nullptr, nullptr, nullptr, qb, DM, DM, T);
        gemm_wmma_bf16<<<dim3(nBlk, DM / BM, B), blk, 0, stream>>>(
            wk, out, nullptr, nullptr, nullptr, kb, DM, DM, T);
        gemm_wmma_bf16<<<dim3(nBlk, DM / BM, B), blk, 0, stream>>>(
            wv, out, nullptr, nullptr, nullptr, vb, DM, DM, T);

        dal_attn_relu<<<(unsigned)((N1 + 255) / 256), blk, 0, stream>>>(
            qb, kb, vb, hb, dils[l], T, (long long)N1);

        // bottle: bias + residual(out) + mask, written in place to out
        gemm_wmma_bf16<<<dim3(nBlk, DM / BM, B), blk, 0, stream>>>(
            bw, hb, bb, out, mask, out, DM, DM, T);
    }

    // bottle6: NC x DM GEMM + bias, * mask -> d_out
    gemm_wmma_bf16<<<dim3(nBlk, (NC + BM - 1) / BM, B), blk, 0, stream>>>(
        b6w, out, b6b, nullptr, mask, (float*)d_out, NC, DM, T);
}
